// ControlLoRACrossAttnProcessor_53987738911664
// MI455X (gfx1250) — matrix operand order repaired and checked by reading, not compiled, not run on hardware
//
#include <hip/hip_runtime.h>
#include <cmath>

// ---------------- problem constants ----------------
#define BB 2
#define SS 2048
#define DD 1024
#define HH 8
#define HDD 128
#define BS (BB * SS)          // 4096
#define RANK 4

typedef __attribute__((ext_vector_type(16))) __bf16 v16bf;
typedef __attribute__((ext_vector_type(8)))  __bf16 v8bf;
typedef __attribute__((ext_vector_type(8)))  float  v8f;
typedef __attribute__((ext_vector_type(4)))  float  v4f;
typedef __attribute__((ext_vector_type(4)))  unsigned int u32x4;
typedef __attribute__((ext_vector_type(8)))  int i32x8;
typedef __attribute__((ext_vector_type(4)))  int i32x4;

#ifndef __has_builtin
#define __has_builtin(x) 0
#endif

#if __has_builtin(__builtin_amdgcn_tensor_load_to_lds)
#define HAVE_TDM 1
#if defined(__clang_major__) && __clang_major__ >= 23
#define TDM_SIX_ARGS 1
#else
#define TDM_SIX_ARGS 0
#endif
#else
#define HAVE_TDM 0
#endif

#if __has_builtin(__builtin_amdgcn_s_wait_tensorcnt)
#define WAIT_TENSORCNT(n) __builtin_amdgcn_s_wait_tensorcnt((short)(n))
#else
#define WAIT_TENSORCNT(n) asm volatile("s_wait_tensorcnt %0" ::"n"(n) : "memory")
#endif

#define WMMA_BF16(a, b, c) \
  __builtin_amdgcn_wmma_f32_16x16x32_bf16(false, (a), false, (b), (short)0, (c), false, false)

// ---------------- fragment loaders (per ISA 7.12.2 VGPR layouts) ----------
// A 16x32 bf16: lane L: M=L&15, g=L>>4; elems 0..7 <-> K=8g+e, 8..15 <-> K=16+8g+(e-8)
__device__ __forceinline__ v16bf load_a16_bf(const __bf16* p, int g) {
  v8bf lo = *(const v8bf*)(p + 8 * g);
  v8bf hi = *(const v8bf*)(p + 16 + 8 * g);
  return __builtin_shufflevector(lo, hi, 0,1,2,3,4,5,6,7,8,9,10,11,12,13,14,15);
}
// B 32x16 bf16: lane L: N=L&15, g=L>>4; elems 0..7 <-> K=8g+e, 8..15 <-> K=16+8g+(e-8)
__device__ __forceinline__ v16bf load_b16_bf(const __bf16* p, int g) {
  v8bf lo = *(const v8bf*)(p + 8 * g);
  v8bf hi = *(const v8bf*)(p + 16 + 8 * g);
  return __builtin_shufflevector(lo, hi, 0,1,2,3,4,5,6,7,8,9,10,11,12,13,14,15);
}

__device__ __forceinline__ float redmax16(float x) {
  x = fmaxf(x, __shfl_xor(x, 1, 32));
  x = fmaxf(x, __shfl_xor(x, 2, 32));
  x = fmaxf(x, __shfl_xor(x, 4, 32));
  x = fmaxf(x, __shfl_xor(x, 8, 32));
  return x;
}
__device__ __forceinline__ float redsum16(float x) {
  x += __shfl_xor(x, 1, 32);
  x += __shfl_xor(x, 2, 32);
  x += __shfl_xor(x, 4, 32);
  x += __shfl_xor(x, 8, 32);
  return x;
}

#if HAVE_TDM
// Issue a 2D TDM load (bf16 elements) of tile_w x tile_h from global into LDS,
// with LDS row padding (pad encodings per D# group1: interval 0=2DW..7=256DW,
// amount 0=1DW..127=128DW).
__device__ __forceinline__ void tdm_load_2d(const void* gp, unsigned lds_off,
                                            int tensor_w, int tensor_h,
                                            int tile_w, int tile_h,
                                            int row_stride, int pad_int_enc,
                                            int pad_amt_enc) {
  unsigned long long ga = (unsigned long long)(size_t)gp;
  u32x4 g0;
  g0[0] = 1u;                                           // count=1, user mode
  g0[1] = lds_off;                                      // lds_addr
  g0[2] = (unsigned)(ga & 0xffffffffu);                 // global_addr lo
  g0[3] = (unsigned)((ga >> 32) & 0x1ffffffu) | (2u << 30);  // addr hi | type=2
  i32x8 g1;
  g1[0] = (int)((1u << 16) | (1u << 20) | ((unsigned)pad_int_enc << 22) |
                ((unsigned)pad_amt_enc << 25));         // 2B elems, pad enable
  g1[1] = (int)(((unsigned)tensor_w & 0xffffu) << 16);  // tensor_dim0 lo16
  g1[2] = (int)((((unsigned)tensor_w >> 16) & 0xffffu) |
                (((unsigned)tensor_h & 0xffffu) << 16));
  g1[3] = (int)((((unsigned)tensor_h >> 16) & 0xffffu) |
                (((unsigned)tile_w & 0xffffu) << 16));  // tile_dim0
  g1[4] = (int)((unsigned)tile_h & 0xffffu);            // tile_dim1, tile_dim2=0
  g1[5] = row_stride;                                   // tensor_dim0_stride lo
  g1[6] = 0;
  g1[7] = 0;
  i32x4 z4 = {0, 0, 0, 0};
#if TDM_SIX_ARGS
  i32x8 z8 = {0, 0, 0, 0, 0, 0, 0, 0};
  __builtin_amdgcn_tensor_load_to_lds(g0, g1, z4, z4, z8, 0);
#else
  __builtin_amdgcn_tensor_load_to_lds(g0, g1, z4, z4, 0);
#endif
}
#endif  // HAVE_TDM

// ---------------- kernel: fp32 -> bf16 bulk convert ----------------
__global__ __launch_bounds__(256) void k_cvt_bf16(const float* __restrict__ in,
                                                  __bf16* __restrict__ out, int n) {
  int i = (blockIdx.x * 256 + threadIdx.x) * 8;
  if (i >= n) return;
  v4f a = ((const v4f*)(in + i))[0];
  v4f b = ((const v4f*)(in + i))[1];
  v8bf r;
#pragma unroll
  for (int j = 0; j < 4; ++j) {
    r[j] = (__bf16)a[j];
    r[4 + j] = (__bf16)b[j];
  }
  *(v8bf*)(out + i) = r;
}

// ---------------- kernel: LoRA down projection (rank 4) ----------------
__global__ __launch_bounds__(256) void k_lora_down(const float* __restrict__ ctrl,
                                                   const float* __restrict__ ldown,
                                                   float* __restrict__ tmp) {
  int row = blockIdx.x * 256 + threadIdx.x;
  if (row >= BS) return;
  const v4f* c = (const v4f*)(ctrl + (size_t)row * DD);
  const v4f* w0 = (const v4f*)(ldown + 0 * DD);
  const v4f* w1 = (const v4f*)(ldown + 1 * DD);
  const v4f* w2 = (const v4f*)(ldown + 2 * DD);
  const v4f* w3 = (const v4f*)(ldown + 3 * DD);
  float a0 = 0.f, a1 = 0.f, a2 = 0.f, a3 = 0.f;
#pragma unroll 4
  for (int i = 0; i < DD / 4; ++i) {
    v4f x = c[i];
    v4f y0 = w0[i], y1 = w1[i], y2 = w2[i], y3 = w3[i];
#pragma unroll
    for (int j = 0; j < 4; ++j) {
      a0 += x[j] * y0[j];
      a1 += x[j] * y1[j];
      a2 += x[j] * y2[j];
      a3 += x[j] * y3[j];
    }
  }
  v4f out = {a0, a1, a2, a3};
  ((v4f*)tmp)[row] = out;
}

// ---------------- kernel: QKV projection, 32x32 per wave ----------------
// grid = (BS/32, DD/32, 3). Q,K -> [B,H,S,HD] bf16 (Q pre-scaled 1/sqrt(HD));
// V -> transposed [B,H,HD,S] bf16.
__global__ __launch_bounds__(32) void k_qkv(const __bf16* __restrict__ Xb,
                                            const __bf16* __restrict__ Wqb,
                                            const __bf16* __restrict__ Wkb,
                                            const __bf16* __restrict__ Wvb,
                                            __bf16* __restrict__ Qo,
                                            __bf16* __restrict__ Ko,
                                            __bf16* __restrict__ Vto) {
  int lane = threadIdx.x;
  int nlo = lane & 15;
  int g = lane >> 4;
  int z = blockIdx.z;
  const __bf16* W = (z == 0) ? Wqb : (z == 1) ? Wkb : Wvb;

  int r0 = blockIdx.x * 32;
  int c0 = blockIdx.y * 32;
  const __bf16* arow0 = Xb + (size_t)(r0 + nlo) * DD;
  const __bf16* arow1 = arow0 + (size_t)16 * DD;
  const __bf16* brow0 = W + (size_t)(c0 + nlo) * DD;
  const __bf16* brow1 = brow0 + (size_t)16 * DD;

  v8f acc00 = {}, acc01 = {}, acc10 = {}, acc11 = {};
#pragma unroll 4
  for (int k0 = 0; k0 < DD; k0 += 32) {
    __builtin_prefetch(arow0 + k0 + 128, 0, 3);
    __builtin_prefetch(brow0 + k0 + 128, 0, 3);
    v16bf a0 = load_a16_bf(arow0 + k0, g);
    v16bf a1 = load_a16_bf(arow1 + k0, g);
    v16bf b0 = load_b16_bf(brow0 + k0, g);
    v16bf b1 = load_b16_bf(brow1 + k0, g);
    acc00 = WMMA_BF16(a0, b0, acc00);
    acc01 = WMMA_BF16(a0, b1, acc01);
    acc10 = WMMA_BF16(a1, b0, acc10);
    acc11 = WMMA_BF16(a1, b1, acc11);
  }

  const float qscale = 0.08838834764831845f;  // 1/sqrt(128)
#pragma unroll
  for (int ti = 0; ti < 2; ++ti) {
#pragma unroll
    for (int tj = 0; tj < 2; ++tj) {
      v8f acc = (ti == 0) ? (tj == 0 ? acc00 : acc01) : (tj == 0 ? acc10 : acc11);
      int e = c0 + 16 * tj + nlo;
      int h = e >> 7;
      int hd = e & 127;
#pragma unroll
      for (int r = 0; r < 8; ++r) {
        int Mrow = r0 + 16 * ti + r + 8 * g;  // D layout: M = vgpr + 8*(lane>=16)
        int bb = Mrow >> 11;
        int ss = Mrow & 2047;
        float v = acc[r];
        if (z == 0) {
          Qo[(((size_t)bb * HH + h) * SS + ss) * HDD + hd] = (__bf16)(v * qscale);
        } else if (z == 1) {
          Ko[(((size_t)bb * HH + h) * SS + ss) * HDD + hd] = (__bf16)v;
        } else {
          Vto[(((size_t)bb * HH + h) * HDD + hd) * SS + ss] = (__bf16)v;
        }
      }
    }
  }
}

// ---------------- kernel: flash attention ----------------
// grid = (S/64, H, B), 4 waves/block, 16 query rows per wave.
// Wave 0 drives double-buffered TDM loads of the 32-key K tile (32x128) and
// V^T tile (128x32) into padded LDS; all 4 waves consume them via WMMA.
#define KSTRIDE 136   // 128 halves + 8 pad (TDM: pad 4 DW per 64 DW)
#define VSTRIDE 40    // 32 halves + 8 pad  (TDM: pad 4 DW per 16 DW)
#define PSTRIDE 40

__global__ __launch_bounds__(128) void k_flash_attn(const __bf16* __restrict__ Q,
                                                    const __bf16* __restrict__ K,
                                                    const __bf16* __restrict__ Vt,
                                                    __bf16* __restrict__ Aout) {
#if HAVE_TDM
  __shared__ __bf16 Klds[2][32 * KSTRIDE];
  __shared__ __bf16 Vlds[2][128 * VSTRIDE];
#endif
  __shared__ __bf16 Plds[4][16 * PSTRIDE];
  const float LOG2E = 1.4426950408889634f;

  int tid = threadIdx.x;
  int lane = tid & 31;
  int wv = tid >> 5;
  int nlo = lane & 15;
  int g = lane >> 4;
  int q0 = blockIdx.x * 64 + wv * 16;
  int h = blockIdx.y;
  int b = blockIdx.z;
  int bh = b * HH + h;

  const __bf16* qrow = Q + ((size_t)bh * SS + q0 + nlo) * HDD;
  v16bf qa[4];
#pragma unroll
  for (int c = 0; c < 4; ++c) qa[c] = load_a16_bf(qrow + 32 * c, g);

  v8f O[8];
#pragma unroll
  for (int c = 0; c < 8; ++c) O[c] = (v8f){};
  float m[8], l[8];
#pragma unroll
  for (int r = 0; r < 8; ++r) { m[r] = -1e30f; l[r] = 0.f; }

#if HAVE_TDM
  const __bf16* kbase = K + (size_t)bh * SS * HDD;
  const __bf16* vbase = Vt + (size_t)bh * HDD * SS;
  if (wv == 0) {
    tdm_load_2d(kbase, (unsigned)(size_t)(void*)&Klds[0][0], HDD, SS, HDD, 32, HDD, 5, 3);
    tdm_load_2d(vbase, (unsigned)(size_t)(void*)&Vlds[0][0], SS, HDD, 32, 128, SS, 3, 3);
  }
#endif

  for (int j = 0; j < SS; j += 32) {
#if HAVE_TDM
    int cur = (j >> 5) & 1;
    if (wv == 0) {
      if (j + 32 < SS) {
        tdm_load_2d(kbase + (size_t)(j + 32) * HDD,
                    (unsigned)(size_t)(void*)&Klds[cur ^ 1][0], HDD, SS, HDD, 32, HDD, 5, 3);
        tdm_load_2d(vbase + (j + 32),
                    (unsigned)(size_t)(void*)&Vlds[cur ^ 1][0], SS, HDD, 32, 128, SS, 3, 3);
        WAIT_TENSORCNT(2);   // current pair complete; next pair in flight
      } else {
        WAIT_TENSORCNT(0);
      }
    }
    __syncthreads();
    const __bf16* kt = &Klds[cur][0];
    const __bf16* vt = &Vlds[cur][0];
#endif

    v8f s0 = {}, s1 = {};
#if HAVE_TDM
#pragma unroll
    for (int c = 0; c < 4; ++c) {
      s0 = WMMA_BF16(qa[c], load_b16_bf(kt + (size_t)nlo * KSTRIDE + 32 * c, g), s0);
      s1 = WMMA_BF16(qa[c], load_b16_bf(kt + (size_t)(16 + nlo) * KSTRIDE + 32 * c, g), s1);
    }
#else
    const __bf16* krow0 = K + ((size_t)bh * SS + j + nlo) * HDD;
    const __bf16* krow1 = krow0 + 16 * HDD;
#pragma unroll
    for (int c = 0; c < 4; ++c) {
      s0 = WMMA_BF16(qa[c], load_b16_bf(krow0 + 32 * c, g), s0);
      s1 = WMMA_BF16(qa[c], load_b16_bf(krow1 + 32 * c, g), s1);
    }
#endif

    // online softmax; N (keys) lives across the 16-lane group, M = vgpr + 8g
#pragma unroll
    for (int r = 0; r < 8; ++r) {
      float s0r = s0[r], s1r = s1[r];
      float mt = redmax16(fmaxf(s0r, s1r));
      float mn = fmaxf(m[r], mt);
      float alpha = exp2f((m[r] - mn) * LOG2E);
      float p0 = exp2f((s0r - mn) * LOG2E);
      float p1 = exp2f((s1r - mn) * LOG2E);
      float rs = redsum16(p0 + p1);
      l[r] = l[r] * alpha + rs;
      m[r] = mn;
#pragma unroll
      for (int c = 0; c < 8; ++c) O[c][r] *= alpha;
      int M = r + 8 * g;
      Plds[wv][M * PSTRIDE + nlo] = (__bf16)p0;
      Plds[wv][M * PSTRIDE + 16 + nlo] = (__bf16)p1;
    }

    v16bf pa = load_a16_bf(&Plds[wv][nlo * PSTRIDE], g);
#pragma unroll
    for (int c = 0; c < 8; ++c) {
#if HAVE_TDM
      O[c] = WMMA_BF16(pa, load_b16_bf(vt + (size_t)(c * 16 + nlo) * VSTRIDE, g), O[c]);
#else
      const __bf16* vrow = Vt + ((size_t)bh * HDD + c * 16 + nlo) * SS;
      O[c] = WMMA_BF16(pa, load_b16_bf(vrow + j, g), O[c]);
#endif
    }
#if HAVE_TDM
    __syncthreads();  // all waves done with buffers before next TDM overwrite
#endif
  }

  // normalize and write attn in [B,S,D] bf16 (heads re-concatenated)
#pragma unroll
  for (int c = 0; c < 8; ++c) {
#pragma unroll
    for (int r = 0; r < 8; ++r) {
      int srow = q0 + r + 8 * g;
      float val = O[c][r] / l[r];
      Aout[((size_t)(b * SS + srow)) * DD + h * HDD + c * 16 + nlo] = (__bf16)val;
    }
  }
}

// ---------------- kernel: output projection + bias + LoRA, 32x32/wave -----
__global__ __launch_bounds__(32) void k_out_proj(const __bf16* __restrict__ attn,
                                                 const __bf16* __restrict__ Wob,
                                                 const float* __restrict__ bo,
                                                 const float* __restrict__ lup,
                                                 const float* __restrict__ tmp,
                                                 float* __restrict__ out) {
  int lane = threadIdx.x;
  int nlo = lane & 15;
  int g = lane >> 4;
  int r0 = blockIdx.x * 32;
  int c0 = blockIdx.y * 32;
  const __bf16* arow0 = attn + (size_t)(r0 + nlo) * DD;
  const __bf16* arow1 = arow0 + (size_t)16 * DD;
  const __bf16* brow0 = Wob + (size_t)(c0 + nlo) * DD;
  const __bf16* brow1 = brow0 + (size_t)16 * DD;

  v8f acc00 = {}, acc01 = {}, acc10 = {}, acc11 = {};
#pragma unroll 4
  for (int k0 = 0; k0 < DD; k0 += 32) {
    __builtin_prefetch(arow0 + k0 + 128, 0, 3);
    __builtin_prefetch(brow0 + k0 + 128, 0, 3);
    v16bf a0 = load_a16_bf(arow0 + k0, g);
    v16bf a1 = load_a16_bf(arow1 + k0, g);
    v16bf b0 = load_b16_bf(brow0 + k0, g);
    v16bf b1 = load_b16_bf(brow1 + k0, g);
    acc00 = WMMA_BF16(a0, b0, acc00);
    acc01 = WMMA_BF16(a0, b1, acc01);
    acc10 = WMMA_BF16(a1, b0, acc10);
    acc11 = WMMA_BF16(a1, b1, acc11);
  }

  int e0 = c0 + nlo;
  int e1 = c0 + 16 + nlo;
  float bias0 = bo[e0], bias1 = bo[e1];
  v4f up0 = ((const v4f*)lup)[e0];
  v4f up1 = ((const v4f*)lup)[e1];
#pragma unroll
  for (int ti = 0; ti < 2; ++ti) {
#pragma unroll
    for (int r = 0; r < 8; ++r) {
      int Mrow = r0 + 16 * ti + r + 8 * g;
      v4f t = ((const v4f*)tmp)[Mrow];
      float l0 = t[0] * up0[0] + t[1] * up0[1] + t[2] * up0[2] + t[3] * up0[3];
      float l1 = t[0] * up1[0] + t[1] * up1[1] + t[2] * up1[2] + t[3] * up1[3];
      v8f a0 = (ti == 0) ? acc00 : acc10;
      v8f a1 = (ti == 0) ? acc01 : acc11;
      out[(size_t)Mrow * DD + e0] = a0[r] + bias0 + l0;
      out[(size_t)Mrow * DD + e1] = a1[r] + bias1 + l1;
    }
  }
}

// ---------------- launch ----------------
extern "C" void kernel_launch(void* const* d_in, const int* in_sizes, int n_in,
                              void* d_out, int out_size, void* d_ws, size_t ws_size,
                              hipStream_t stream) {
  const float* hidden = (const float*)d_in[0];
  const float* control = (const float*)d_in[1];
  const float* Wq = (const float*)d_in[2];
  const float* Wk = (const float*)d_in[3];
  const float* Wv = (const float*)d_in[4];
  const float* Wo = (const float*)d_in[5];
  const float* bo = (const float*)d_in[6];
  const float* ldown = (const float*)d_in[7];
  const float* lup = (const float*)d_in[8];
  float* out = (float*)d_out;
  (void)in_sizes; (void)n_in; (void)out_size; (void)ws_size;

  // workspace (bf16 elements): Q | K | Vt | Xbf (reused as attn) | W*bf, then tmp f32
  const size_t QE = (size_t)BB * HH * SS * HDD;  // 4,194,304
  const size_t WE = (size_t)DD * DD;             // 1,048,576
  __bf16* qws = (__bf16*)d_ws;
  __bf16* kws = qws + QE;
  __bf16* vtws = qws + 2 * QE;
  __bf16* xbf = qws + 3 * QE;       // consumed by k_qkv ...
  __bf16* attnws = xbf;             // ... then reused for attention output
  __bf16* wqbf = qws + 4 * QE;
  __bf16* wkbf = wqbf + WE;
  __bf16* wvbf = wqbf + 2 * WE;
  __bf16* wobf = wqbf + 3 * WE;
  float* tmpws = (float*)(wqbf + 4 * WE);

  const int NX = BB * SS * DD;     // 4,194,304
  const int NW = DD * DD;          // 1,048,576
  k_cvt_bf16<<<dim3(NX / (8 * 256)), dim3(256), 0, stream>>>(hidden, xbf, NX);
  k_cvt_bf16<<<dim3(NW / (8 * 256)), dim3(256), 0, stream>>>(Wq, wqbf, NW);
  k_cvt_bf16<<<dim3(NW / (8 * 256)), dim3(256), 0, stream>>>(Wk, wkbf, NW);
  k_cvt_bf16<<<dim3(NW / (8 * 256)), dim3(256), 0, stream>>>(Wv, wvbf, NW);
  k_cvt_bf16<<<dim3(NW / (8 * 256)), dim3(256), 0, stream>>>(Wo, wobf, NW);

  k_lora_down<<<dim3(BS / 256), dim3(256), 0, stream>>>(control, ldown, tmpws);

  k_qkv<<<dim3(BS / 32, DD / 32, 3), dim3(32), 0, stream>>>(
      xbf, wqbf, wkbf, wvbf, qws, kws, vtws);

  k_flash_attn<<<dim3(SS / 64, HH, BB), dim3(128), 0, stream>>>(
      qws, kws, vtws, attnws);

  k_out_proj<<<dim3(BS / 32, DD / 32), dim3(32), 0, stream>>>(
      attnws, wobf, bo, lup, tmpws, out);
}
